// Decoder_3513283248319
// MI455X (gfx1250) — compile-verified, matrix operand order checked by reading, not run on hardware
//
#include <hip/hip_runtime.h>

// ---------------------------------------------------------------------------
// Decoder w/ Bahdanau attention + GRU, CDNA5 (gfx1250) bf16 WMMA implementation
// B=64, S=256, EMB=DEC=1024, 2*ENC=1024, GIN=2048, 3*DEC=3072
// Per step (4 kernels): [gh|g] GEMM -> attention -> gi GEMM -> gates(+sel)
// ---------------------------------------------------------------------------

typedef __bf16 bf16_t;
typedef __attribute__((ext_vector_type(16))) __bf16 v16bf;
typedef __attribute__((ext_vector_type(8)))  float  v8f;

#define B_    64
#define S_    256
#define EMB_  1024
#define DEC_  1024
#define ENC2_ 1024
#define GIN_  2048
#define H3_   3072
#define H4_   4096   // 3*DEC (gh) + DEC (g) fused output width

// ---------------------------------------------------------------------------
// Threefry2x32 (JAX PRNG) to reproduce tf_flags = uniform(key(1),(S-1,)) < 0.5
// ---------------------------------------------------------------------------
__device__ __forceinline__ unsigned rotl32(unsigned x, int r) {
    return (x << r) | (x >> (32 - r));
}

__device__ __forceinline__ void threefry2x32(unsigned k0, unsigned k1,
                                             unsigned& x0, unsigned& x1) {
    const unsigned ks2 = k0 ^ k1 ^ 0x1BD11BDAu;
    const int r0[4] = {13, 15, 26, 6};
    const int r1[4] = {17, 29, 16, 24};
    x0 += k0; x1 += k1;
#pragma unroll
    for (int i = 0; i < 4; ++i) { x0 += x1; x1 = rotl32(x1, r0[i]); x1 ^= x0; }
    x0 += k1; x1 += ks2 + 1u;
#pragma unroll
    for (int i = 0; i < 4; ++i) { x0 += x1; x1 = rotl32(x1, r1[i]); x1 ^= x0; }
    x0 += ks2; x1 += k0 + 2u;
#pragma unroll
    for (int i = 0; i < 4; ++i) { x0 += x1; x1 = rotl32(x1, r0[i]); x1 ^= x0; }
    x0 += k0; x1 += k1 + 3u;
#pragma unroll
    for (int i = 0; i < 4; ++i) { x0 += x1; x1 = rotl32(x1, r1[i]); x1 ^= x0; }
    x0 += k1; x1 += ks2 + 4u;
#pragma unroll
    for (int i = 0; i < 4; ++i) { x0 += x1; x1 = rotl32(x1, r0[i]); x1 ^= x0; }
    x0 += ks2; x1 += k0 + 5u;
}

__global__ void tf_flags_kernel(unsigned* flags) {
    int i = threadIdx.x;                       // 0..127
    if (i >= 128) return;
    // counts 0..254 padded to 256: x0 = counts[0..127], x1 = counts[128..255]
    unsigned x0 = (unsigned)i;
    unsigned x1 = (i < 127) ? (unsigned)(128 + i) : 0u;
    threefry2x32(0u, 1u, x0, x1);              // key(1) -> (0,1)
    float f0 = __uint_as_float((x0 >> 9) | 0x3f800000u) - 1.0f;
    float f1 = __uint_as_float((x1 >> 9) | 0x3f800000u) - 1.0f;
    flags[i] = (f0 < 0.5f) ? 1u : 0u;
    if (i < 127) flags[128 + i] = (f1 < 0.5f) ? 1u : 0u;
    if (i == 127) flags[255] = 0u;             // unused slot, keep deterministic
}

// ---------------------------------------------------------------------------
// Prep kernels
// ---------------------------------------------------------------------------
__global__ void conv_bf16_kernel(const float* __restrict__ src,
                                 bf16_t* __restrict__ dst, int n) {
    int idx = blockIdx.x * blockDim.x + threadIdx.x;
    if (idx < n) dst[idx] = (bf16_t)src[idx];
}

// src [K,N] row-major  ->  dst [N,K] row-major (transposed + bf16)
__global__ void convT_bf16_kernel(const float* __restrict__ src,
                                  bf16_t* __restrict__ dst, int K, int N) {
    int idx = blockIdx.x * blockDim.x + threadIdx.x;
    if (idx < K * N) {
        int n = idx / K, k = idx % K;
        dst[idx] = (bf16_t)src[(size_t)k * N + n];
    }
}

// h0 = hidden.T  (hidden is [DEC, B])
__global__ void init_h_kernel(const float* __restrict__ hidden,
                              float* __restrict__ h, bf16_t* __restrict__ h_bf) {
    int idx = blockIdx.x * blockDim.x + threadIdx.x;   // 64*1024
    int b = idx >> 10, d = idx & 1023;
    float val = hidden[(size_t)d * B_ + b];
    h[idx] = val;
    h_bf[idx] = (bf16_t)val;
}

// outs[:,0,:] = input[:,0,:]  and step-1 selection (teach == prev == input0)
__global__ void out0_kernel(const float* __restrict__ input,
                            float* __restrict__ outs,
                            bf16_t* __restrict__ x_bf) {
    int idx = blockIdx.x * blockDim.x + threadIdx.x;   // 64*1024
    int b = idx >> 10, d = idx & 1023;
    float val = input[(size_t)b * S_ * EMB_ + d];
    outs[(size_t)b * S_ * DEC_ + d] = val;
    x_bf[(size_t)b * GIN_ + d] = (bf16_t)val;
}

// ---------------------------------------------------------------------------
// bf16 WMMA GEMM:  C[M,N] = A[M,K] * B^T   (B given as [N,K] row-major)
// grid = (N/32, M/64), block = 64 (2 waves, 16 N-cols per wave).
// Each wave loads one B fragment per K-chunk and reuses it across four
// M=16 tiles held in registers (4 x v8f accumulators) -> weights stream
// through L2 exactly once per GEMM.
// v_wmma_f32_16x16x32_bf16 operand layout (wave32):
//   lanes 0-15 : row = lane,    elements = K {k..k+7} then {k+16..k+23}
//   lanes 16-31: row = lane-16, elements = K {k+8..k+15} then {k+24..k+31}
// identical indexing for A ([M,K] rows) and B ([N,K] rows).
// ---------------------------------------------------------------------------
template <bool A_F32, bool C_BF16>
__global__ __launch_bounds__(64)
void wmma_gemm_kernel(const void* __restrict__ Aptr,
                      const bf16_t* __restrict__ Bmat,
                      void* __restrict__ Cptr, int M, int N, int K) {
    const int lane = threadIdx.x & 31;
    const int wave = threadIdx.x >> 5;
    const int r    = lane & 15;
    const int half = lane >> 4;
    const int klo  = half * 8;
    const int n0   = (blockIdx.x * 2 + wave) << 4;
    const int m0   = blockIdx.y << 6;            // 4 tiles of 16 rows

    const bf16_t* Brow = Bmat + (size_t)(n0 + r) * K;

    v8f acc0 = {0.f,0.f,0.f,0.f,0.f,0.f,0.f,0.f};
    v8f acc1 = acc0, acc2 = acc0, acc3 = acc0;
    union Frag { v16bf v; uint4 q[2]; } a, b;

    if constexpr (A_F32) {
        const float* Abase = (const float*)Aptr + (size_t)(m0 + r) * K;
        for (int k = 0; k < K; k += 32) {
            b.q[0] = *(const uint4*)(Brow + k + klo);
            b.q[1] = *(const uint4*)(Brow + k + klo + 16);
#pragma unroll
            for (int mi = 0; mi < 4; ++mi) {
                const float* Arow = Abase + (size_t)(mi * 16) * K;
                float fa[16];
                *(float4*)(fa + 0)  = *(const float4*)(Arow + k + klo);
                *(float4*)(fa + 4)  = *(const float4*)(Arow + k + klo + 4);
                *(float4*)(fa + 8)  = *(const float4*)(Arow + k + klo + 16);
                *(float4*)(fa + 12) = *(const float4*)(Arow + k + klo + 20);
#pragma unroll
                for (int j = 0; j < 16; ++j) a.v[j] = (bf16_t)fa[j];
                v8f& acc = (mi == 0) ? acc0 : (mi == 1) ? acc1
                                            : (mi == 2) ? acc2 : acc3;
                acc = __builtin_amdgcn_wmma_f32_16x16x32_bf16(
                    false, a.v, false, b.v, (short)0, acc, false, false);
            }
        }
    } else {
        const bf16_t* Abase = (const bf16_t*)Aptr + (size_t)(m0 + r) * K;
        for (int k = 0; k < K; k += 32) {
            b.q[0] = *(const uint4*)(Brow + k + klo);
            b.q[1] = *(const uint4*)(Brow + k + klo + 16);
#pragma unroll
            for (int mi = 0; mi < 4; ++mi) {
                const bf16_t* Arow = Abase + (size_t)(mi * 16) * K;
                a.q[0] = *(const uint4*)(Arow + k + klo);
                a.q[1] = *(const uint4*)(Arow + k + klo + 16);
                v8f& acc = (mi == 0) ? acc0 : (mi == 1) ? acc1
                                            : (mi == 2) ? acc2 : acc3;
                acc = __builtin_amdgcn_wmma_f32_16x16x32_bf16(
                    false, a.v, false, b.v, (short)0, acc, false, false);
            }
        }
    }

    // C/D layout: lanes 0-15 -> M = tile + v, col = n0+lane;
    //             lanes 16-31 -> M = tile + 8 + v
    const int col = n0 + r;
#pragma unroll
    for (int mi = 0; mi < 4; ++mi) {
        const v8f& acc = (mi == 0) ? acc0 : (mi == 1) ? acc1
                                          : (mi == 2) ? acc2 : acc3;
        const int mb = m0 + mi * 16 + half * 8;
        if constexpr (C_BF16) {
            bf16_t* C = (bf16_t*)Cptr;
#pragma unroll
            for (int i = 0; i < 8; ++i)
                C[(size_t)(mb + i) * N + col] = (bf16_t)acc[i];
        } else {
            float* C = (float*)Cptr;
#pragma unroll
            for (int i = 0; i < 8; ++i)
                C[(size_t)(mb + i) * N + col] = acc[i];
        }
    }
}

// ---------------------------------------------------------------------------
// Attention: energy = v . tanh(enc_proj[b,s,:] + g[b,:]); masked softmax over s;
// ctx[b,:] = sum_s attn[s] * enc[b,s,:]   -> bf16 into x[:,1024:2048]
// One workgroup (512 thr = 16 waves) per batch row. g has row stride ldg.
// ---------------------------------------------------------------------------
__device__ __forceinline__ float sigmoidf_(float x) {
    return 1.0f / (1.0f + __expf(-x));
}

__global__ __launch_bounds__(512)
void attention_kernel(const bf16_t* __restrict__ enc_proj,  // [B,S,DEC] bf16
                      const float* __restrict__ g,          // [B,DEC] stride ldg
                      int ldg,
                      const float* __restrict__ vvec,       // [DEC]
                      const float* __restrict__ enc,        // [B,S,2ENC] f32
                      const int* __restrict__ lengths,      // [B]
                      bf16_t* __restrict__ x_bf) {          // [B,GIN]
    __shared__ float sg[DEC_];
    __shared__ float sv[DEC_];
    __shared__ float se[S_];
    __shared__ float red[S_];

    const int b = blockIdx.x;
    const int tid = threadIdx.x;
    const int lane = tid & 31;
    const int wave = tid >> 5;                  // 0..15

    for (int d = tid; d < DEC_; d += 512) {
        sg[d] = g[(size_t)b * ldg + d];
        sv[d] = vvec[d];
    }
    __syncthreads();

    const int len = lengths[b];

    // energies: wave w handles s = w, w+16, ...
    for (int s = wave; s < S_; s += 16) {
        const bf16_t* row = enc_proj + ((size_t)(b * S_ + s)) * DEC_;
        float sum = 0.f;
        for (int d0 = lane * 8; d0 < DEC_; d0 += 32 * 8) {
            union { uint4 q; bf16_t e[8]; } u;
            u.q = *(const uint4*)(row + d0);
#pragma unroll
            for (int j = 0; j < 8; ++j)
                sum += tanhf((float)u.e[j] + sg[d0 + j]) * sv[d0 + j];
        }
#pragma unroll
        for (int off = 16; off > 0; off >>= 1)
            sum += __shfl_down(sum, off, 32);
        if (lane == 0) se[s] = (s < len) ? sum : -1e9f;
    }
    __syncthreads();

    // masked softmax over 256 entries (first 256 threads own one entry each)
    if (tid < S_) red[tid] = se[tid];
    __syncthreads();
    for (int off = 128; off > 0; off >>= 1) {
        if (tid < off) red[tid] = fmaxf(red[tid], red[tid + off]);
        __syncthreads();
    }
    float mx = red[0];
    __syncthreads();
    float p = 0.f;
    if (tid < S_) { p = __expf(se[tid] - mx); red[tid] = p; }
    __syncthreads();
    for (int off = 128; off > 0; off >>= 1) {
        if (tid < off) red[tid] += red[tid + off];
        __syncthreads();
    }
    float inv = 1.0f / red[0];
    __syncthreads();
    if (tid < S_) se[tid] = p * inv;
    __syncthreads();

    // ctx[b, d..d+1] for d = tid*2
    const float* encb = enc + (size_t)b * S_ * ENC2_;
    const int d = tid * 2;
    float cx = 0.f, cy = 0.f;
    for (int s = 0; s < S_; ++s) {
        float aw = se[s];
        float2 e2 = *(const float2*)(encb + (size_t)s * ENC2_ + d);
        cx += aw * e2.x;
        cy += aw * e2.y;
    }
    bf16_t* xo = x_bf + (size_t)b * GIN_ + DEC_ + d;
    xo[0] = (bf16_t)cx;
    xo[1] = (bf16_t)cy;
}

// ---------------------------------------------------------------------------
// GRU gates + state update; writes h (f32), h_bf16, outs[:,t,:], and the
// NEXT step's teacher-forcing selection x[:,0:1024] (sel = tf[t]?input[:,t]:h_new)
// gh rows have stride H4_ (fused [gh|g] buffer).
// ---------------------------------------------------------------------------
__global__ void gate_kernel(const float* __restrict__ gi,    // [B,3*DEC]
                            const float* __restrict__ ghg,   // [B,H4_]
                            const float* __restrict__ b_ih,
                            const float* __restrict__ b_hh,
                            float* __restrict__ h,
                            bf16_t* __restrict__ h_bf,
                            float* __restrict__ outs,
                            const float* __restrict__ input,
                            const unsigned* __restrict__ flags,
                            bf16_t* __restrict__ x_bf, int t) {
    int idx = blockIdx.x * blockDim.x + threadIdx.x;   // 64*1024
    int b = idx >> 10, j = idx & 1023;
    const float* gib = gi + (size_t)b * H3_;
    const float* ghb = ghg + (size_t)b * H4_;
    float rg = sigmoidf_(gib[j] + b_ih[j] + ghb[j] + b_hh[j]);
    float zg = sigmoidf_(gib[DEC_ + j] + b_ih[DEC_ + j] +
                         ghb[DEC_ + j] + b_hh[DEC_ + j]);
    float ng = tanhf(gib[2 * DEC_ + j] + b_ih[2 * DEC_ + j] +
                     rg * (ghb[2 * DEC_ + j] + b_hh[2 * DEC_ + j]));
    float hv = h[idx];
    float hn = (1.0f - zg) * ng + zg * hv;
    h[idx] = hn;
    h_bf[idx] = (bf16_t)hn;
    outs[((size_t)b * S_ + t) * DEC_ + j] = hn;
    // selection feeding step t+1 (flags[255]==0, write is dead on last step)
    float selv = flags[t] ? input[((size_t)b * S_ + t) * EMB_ + j] : hn;
    x_bf[(size_t)b * GIN_ + j] = (bf16_t)selv;
}

// ---------------------------------------------------------------------------
// Host launcher
// ---------------------------------------------------------------------------
extern "C" void kernel_launch(void* const* d_in, const int* in_sizes, int n_in,
                              void* d_out, int out_size, void* d_ws,
                              size_t ws_size, hipStream_t stream) {
    const float* input   = (const float*)d_in[0];   // [B,S,EMB]
    const float* hidden  = (const float*)d_in[1];   // [DEC,B]
    const float* enc     = (const float*)d_in[2];   // [B,S,2ENC]
    const int*   lengths = (const int*)d_in[3];     // [B]
    const float* W_enc   = (const float*)d_in[4];   // [2ENC,DEC]
    const float* W_dec   = (const float*)d_in[5];   // [DEC,DEC]
    const float* vvec    = (const float*)d_in[6];   // [DEC]
    const float* W_ih    = (const float*)d_in[7];   // [3DEC,GIN]
    const float* W_hh    = (const float*)d_in[8];   // [3DEC,DEC]
    const float* b_ih    = (const float*)d_in[9];
    const float* b_hh    = (const float*)d_in[10];
    float* outs = (float*)d_out;                    // [B,S,DEC]

    char* ws = (char*)d_ws;
    size_t off = 0;
    auto carve = [&](size_t bytes) -> void* {
        off = (off + 255) & ~(size_t)255;
        void* p = ws + off;
        off += bytes;
        return p;
    };

    unsigned* flags   = (unsigned*)carve(256 * sizeof(unsigned));
    float*    h       = (float*)   carve((size_t)B_ * DEC_ * 4);
    bf16_t*   h_bf    = (bf16_t*)  carve((size_t)B_ * DEC_ * 2);
    bf16_t*   x_bf    = (bf16_t*)  carve((size_t)B_ * GIN_ * 2);
    float*    gi      = (float*)   carve((size_t)B_ * H3_ * 4);
    float*    ghg     = (float*)   carve((size_t)B_ * H4_ * 4);  // [gh | g]
    bf16_t*   WencT   = (bf16_t*)  carve((size_t)ENC2_ * DEC_ * 2);
    bf16_t*   Wcat    = (bf16_t*)  carve((size_t)H4_ * DEC_ * 2); // W_hh ‖ W_dec^T
    bf16_t*   Wih_bf  = (bf16_t*)  carve((size_t)H3_ * GIN_ * 2);
    bf16_t*   encproj = (bf16_t*)  carve((size_t)B_ * S_ * DEC_ * 2);
    (void)ws_size; (void)in_sizes; (void)n_in; (void)out_size;

    // ---- one-time prep (re-done every call; deterministic) ----
    tf_flags_kernel<<<1, 128, 0, stream>>>(flags);

    conv_bf16_kernel<<<(H3_ * GIN_) / 256, 256, 0, stream>>>(W_ih, Wih_bf,
                                                             H3_ * GIN_);
    // Wcat rows 0..3071 = W_hh (already [N,K]); rows 3072..4095 = W_dec^T
    conv_bf16_kernel<<<(H3_ * DEC_) / 256, 256, 0, stream>>>(W_hh, Wcat,
                                                             H3_ * DEC_);
    convT_bf16_kernel<<<(DEC_ * DEC_) / 256, 256, 0, stream>>>(
        W_dec, Wcat + (size_t)H3_ * DEC_, DEC_, DEC_);
    convT_bf16_kernel<<<(ENC2_ * DEC_) / 256, 256, 0, stream>>>(W_enc, WencT,
                                                                ENC2_, DEC_);
    init_h_kernel<<<(B_ * DEC_) / 256, 256, 0, stream>>>(hidden, h, h_bf);
    out0_kernel<<<(B_ * DEC_) / 256, 256, 0, stream>>>(input, outs, x_bf);

    // enc_proj = encoder_outs @ W_enc  (M=16384, N=1024, K=1024), bf16 output
    wmma_gemm_kernel<true, true>
        <<<dim3(ENC2_ / 32, (B_ * S_) / 64), 64, 0, stream>>>(
            enc, WencT, encproj, B_ * S_, DEC_, ENC2_);

    // ---- sequential decode: t = 1 .. S-1 ----
    for (int t = 1; t < S_; ++t) {
        // [gh | g] = h @ [W_hh ‖ W_dec^T]^T   (M=64, N=4096, K=1024)
        wmma_gemm_kernel<false, false>
            <<<dim3(H4_ / 32, 1), 64, 0, stream>>>(
                h_bf, Wcat, ghg, B_, H4_, DEC_);

        // attention -> x[:,1024:2048]  (g = ghg cols 3072.., stride H4_)
        attention_kernel<<<B_, 512, 0, stream>>>(encproj, ghg + H3_, H4_,
                                                 vvec, enc, lengths, x_bf);

        // gi = x @ W_ih^T              (M=64, N=3072, K=2048)
        wmma_gemm_kernel<false, false>
            <<<dim3(H3_ / 32, 1), 64, 0, stream>>>(
                x_bf, Wih_bf, gi, B_, H3_, GIN_);

        // gates + state update + outs[:,t,:] + next-step selection
        gate_kernel<<<(B_ * DEC_) / 256, 256, 0, stream>>>(
            gi, ghg, b_ih, b_hh, h, h_bf, outs, input, flags, x_bf, t);
    }
}
